// CountLoss_66537633349907
// MI455X (gfx1250) — compile-verified, matrix operand order checked by reading, not run on hardware
//
#include <hip/hip_runtime.h>
#include <hip/hip_bf16.h>

typedef __attribute__((ext_vector_type(16))) _Float16 v16h;
typedef __attribute__((ext_vector_type(8)))  float    v8f;

#define NSAMP 32
#define NBINS 256
#define NPIX  (512 * 512)
#define BLOCKS_PER_SAMPLE 32
#define ELEMS_PER_BLOCK (NPIX / BLOCKS_PER_SAMPLE)   // 8192 labels per block
#define THREADS 256
#define VEC_ITERS (ELEMS_PER_BLOCK / (THREADS * 4))  // 8 int4 loads per thread

// ---------------------------------------------------------------------------
// Kernel 0: zero the 32 x 4 u64 presence table (d_ws is not re-cleared by the
// harness between replays, so we must do it ourselves every launch).
// ---------------------------------------------------------------------------
__global__ void zero_presence(unsigned long long* __restrict__ presence) {
    int t = threadIdx.x;
    if (t < NSAMP * 4) presence[t] = 0ull;
}

// ---------------------------------------------------------------------------
// Kernel 1: streaming presence builder. Memory-bound: 32 MiB of int32 labels,
// read as coalesced 16-byte (B128) loads. Labels are in [0,150), so only
// three u64 presence words are reachable (sel = v>>6 in {0,1,2}) -> 3
// branchless conditional ORs per label (~17 VALU/label). Per-thread register
// bitmask, merged with ds_or_b64 LDS atomics, then one global_atomic_or_b64
// per word per block.
// ---------------------------------------------------------------------------
__global__ void presence_kernel(const int* __restrict__ label,
                                unsigned long long* __restrict__ presence) {
    __shared__ unsigned long long smask[3];
    if (threadIdx.x < 3) smask[threadIdx.x] = 0ull;
    __syncthreads();

    const int sample = blockIdx.y;
    const size_t base = (size_t)sample * NPIX + (size_t)blockIdx.x * ELEMS_PER_BLOCK;
    const int4* p = reinterpret_cast<const int4*>(label + base);

    unsigned long long m0 = 0, m1 = 0, m2 = 0;

#define SETBIT(val)                                                   \
    {                                                                 \
        unsigned _v = (unsigned)(val);                                \
        unsigned long long _b = 1ull << (_v & 63u); /* hw-masked */   \
        unsigned _w = _v >> 6;          /* in {0,1,2} for v<192 */    \
        m0 |= (_w == 0u) ? _b : 0ull;                                 \
        m1 |= (_w == 1u) ? _b : 0ull;                                 \
        m2 |= (_w >= 2u) ? _b : 0ull;                                 \
    }

#pragma unroll
    for (int i = 0; i < VEC_ITERS; ++i) {
        if (i + 2 < VEC_ITERS)  // speculative prefetch ahead of the stream
            __builtin_prefetch(&p[threadIdx.x + (i + 2) * THREADS], 0, 0);
        int4 v = p[threadIdx.x + i * THREADS];
        SETBIT(v.x); SETBIT(v.y); SETBIT(v.z); SETBIT(v.w);
    }
#undef SETBIT

    if (m0) atomicOr(&smask[0], m0);
    if (m1) atomicOr(&smask[1], m1);
    if (m2) atomicOr(&smask[2], m2);
    __syncthreads();

    if (threadIdx.x < 3) {
        unsigned long long w = smask[threadIdx.x];
        if (w) atomicOr(&presence[sample * 4 + threadIdx.x], w);
    }
}

// ---------------------------------------------------------------------------
// Kernel 2 (1 block, 64 threads = 2 waves):
//  Phase 1: counts[b] = popcount(presence[b]) computed as a WMMA matvec:
//           D = A(presence bits as fp16 0/1, 16x32 per step) x B(ones) + C,
//           accumulated over 8 K-chunks of 32 bins -> v_wmma_f32_16x16x32_f16.
//           Wave w handles samples [16w, 16w+16).
//           A-fragment layout per ISA 7.12.2 (16-bit A 16x32):
//             lanes 0-15:  e<8 -> K=e,    e>=8 -> K=e+8  (16..23)
//             lanes 16-31: e<8 -> K=e+8,  e>=8 -> K=e+16 (24..31)
//           D layout: lane 0 holds D[0..7][0], lane 16 holds D[8..15][0].
//  Phase 2: wave 0 computes per-sample NLL (log-softmax over 256 classes at
//           index counts[b]) and reduces the mean with wave32 shuffles.
// ---------------------------------------------------------------------------
__global__ void finalize_kernel(const float* __restrict__ cls,
                                const unsigned long long* __restrict__ presence,
                                float* __restrict__ out) {
    __shared__ int counts_s[NSAMP];

    const int tid  = threadIdx.x;
    const int wave = tid >> 5;
    const int lane = tid & 31;
    const int m    = lane & 15;
    const int hi   = lane >> 4;
    const int sample = wave * 16 + m;

    unsigned long long pw0 = presence[sample * 4 + 0];
    unsigned long long pw1 = presence[sample * 4 + 1];
    unsigned long long pw2 = presence[sample * 4 + 2];
    unsigned long long pw3 = presence[sample * 4 + 3];

    v16h bmat;
#pragma unroll
    for (int e = 0; e < 16; ++e) bmat[e] = (_Float16)1.0f;

    v8f c = {};
#pragma unroll
    for (int kc = 0; kc < 8; ++kc) {
        // chunk kc covers bins [32*kc, 32*kc+32) -> word kc/2, bit base (kc&1)*32
        unsigned long long w = (kc < 2) ? pw0 : (kc < 4) ? pw1 : (kc < 6) ? pw2 : pw3;
        const int kbase = (kc & 1) * 32;
        v16h a;
#pragma unroll
        for (int e = 0; e < 16; ++e) {
            int K = e + ((e >= 8) ? 8 : 0) + (hi ? 8 : 0);
            a[e] = (_Float16)(float)((w >> (kbase + K)) & 1ull);
        }
        c = __builtin_amdgcn_wmma_f32_16x16x32_f16(
                /*neg_a=*/false, a, /*neg_b=*/false, bmat,
                /*c_mod=*/(short)0, c, /*reuse_a=*/false, /*reuse_b=*/false);
    }

    if (lane == 0) {
#pragma unroll
        for (int r = 0; r < 8; ++r) counts_s[wave * 16 + r] = (int)(c[r] + 0.5f);
    } else if (lane == 16) {
#pragma unroll
        for (int r = 0; r < 8; ++r) counts_s[wave * 16 + 8 + r] = (int)(c[r] + 0.5f);
    }
    __syncthreads();

    float nll = 0.0f;
    if (tid < NSAMP) {
        const float* row = cls + tid * NBINS;
        float mx = -3.402823466e38f;
        for (int i = 0; i < NBINS; ++i) mx = fmaxf(mx, row[i]);
        float se = 0.0f;
        for (int i = 0; i < NBINS; ++i) se += expf(row[i] - mx);
        int tgt = counts_s[tid];
        tgt = (tgt < 0) ? 0 : ((tgt > NBINS - 1) ? NBINS - 1 : tgt);
        nll = logf(se) + mx - row[tgt];
    }
    if (tid < 32) {
        for (int off = 16; off > 0; off >>= 1)
            nll += __shfl_xor(nll, off, 32);
        if (tid == 0) out[0] = nll / (float)NSAMP;
    }
}

// ---------------------------------------------------------------------------
extern "C" void kernel_launch(void* const* d_in, const int* in_sizes, int n_in,
                              void* d_out, int out_size, void* d_ws, size_t ws_size,
                              hipStream_t stream) {
    (void)in_sizes; (void)n_in; (void)out_size; (void)ws_size;

    const float* cls   = (const float*)d_in[0];   // [32, 256] f32
    const int*   label = (const int*)d_in[1];     // [32, 512, 512] int32
    float* out = (float*)d_out;                   // scalar
    unsigned long long* presence = (unsigned long long*)d_ws;  // [32][4] u64

    zero_presence<<<1, 128, 0, stream>>>(presence);
    presence_kernel<<<dim3(BLOCKS_PER_SAMPLE, NSAMP), THREADS, 0, stream>>>(label, presence);
    finalize_kernel<<<1, 64, 0, stream>>>(cls, presence, out);
}
